// SelfAttention_60138132079095
// MI455X (gfx1250) — matrix operand order repaired and checked by reading, not compiled, not run on hardware
//
#include <hip/hip_runtime.h>

// ---- CDNA5 (gfx1250) wave32 WMMA self-attention pipeline ----
// cvt(fp32->bf16) -> QKV gemm (TDM-staged B) -> flash attention ->
// output projection (TDM-staged B).
// Matmuls: v_wmma_f32_16x16x32_bf16.  B operands are DMA'd into LDS with
// tensor_load_to_lds (double-buffered, TENSORcnt-tracked) and read back as
// padded ds_load_b128 fragments; A operands are register double-buffered
// global b128 loads.

typedef __attribute__((ext_vector_type(16))) __bf16 v16bf;
typedef __attribute__((ext_vector_type(8)))  __bf16 v8bf;
typedef __attribute__((ext_vector_type(8)))  float  v8f;
typedef __attribute__((ext_vector_type(4)))  unsigned int u32x4;
typedef __attribute__((ext_vector_type(8)))  int i32x8;
typedef __attribute__((ext_vector_type(4)))  int i32x4;

#define B_   4
#define N_   1024
#define E_   1024
#define H_   16
#define D_   64
#define BH_  (B_ * H_)

// LDS B-tile: 64 rows x 32 K bf16, row pitch padded 32->40 elems (80 B =
// 5x16 B) by the TDM pad feature -> conflict-reduced, 16B-aligned ds reads.
#define BT_ROWS   64
#define BT_K      32
#define BT_PITCH  40
#define BT_ELEMS  (BT_ROWS * BT_PITCH)   // 2560 elems = 5120 B per buffer

static __device__ inline v8f wmma_bf16(v16bf a, v16bf b, v8f c) {
  return __builtin_amdgcn_wmma_f32_16x16x32_bf16(false, a, false, b, (short)0, c,
                                                 false, false);
}

// A-fragment (16x32, 16-bit): lane m=lane&15 holds K-chunks [g*16,+8) and
// [g*16+8,+8) of row (row0+m).  Source bf16 row-major, K contiguous.
static __device__ inline v16bf load_a_bf16(const __bf16* __restrict__ A, int lda,
                                           int row0, int kbase, int lane) {
  const int m = lane & 15, g = lane >> 4;
  const __bf16* p = A + (size_t)(row0 + m) * lda + kbase;
  v8bf lo = *(const v8bf*)(p + g * 16);
  v8bf hi = *(const v8bf*)(p + g * 16 + 8);
  v16bf a;
#pragma unroll
  for (int i = 0; i < 8; ++i) { a[i] = lo[i]; a[i + 8] = hi[i]; }
  return a;
}

// B-fragment (32x16) from bf16 global "Bt" [Nout,K] row-major.
static __device__ inline v16bf load_bt_bf16(const __bf16* __restrict__ Bt, int ldb,
                                            int col0, int kbase, int lane) {
  const int n = lane & 15, g = lane >> 4;
  return *(const v16bf*)(Bt + (size_t)(col0 + n) * ldb + kbase + g * 16);
}

// B-fragment from the padded LDS tile (row = output column, BT_PITCH pitch).
static __device__ inline v16bf lds_bt(const __bf16* buf, int colt, int lane) {
  const int n = lane & 15, g = lane >> 4;
  const __bf16* p = buf + (size_t)(colt + n) * BT_PITCH + g * 16;
  v8bf lo = *(const v8bf*)p;
  v8bf hi = *(const v8bf*)(p + 8);
  v16bf b;
#pragma unroll
  for (int i = 0; i < 8; ++i) { b[i] = lo[i]; b[i + 8] = hi[i]; }
  return b;
}

// TDM: DMA a 64x32 bf16 tile (row stride = src_stride elems) into LDS at
// lds_byte_off, padding each 64B row with 16B.  D# per ISA 08 section 8.
static __device__ inline void tdm_load_tile(const __bf16* gsrc, unsigned lds_byte_off,
                                            unsigned tdim0, unsigned tdim1,
                                            unsigned src_stride) {
  const unsigned long long ga = (unsigned long long)(uintptr_t)gsrc;
  u32x4 g0;
  g0[0] = 1u;                                        // count=1, user mode
  g0[1] = lds_byte_off;                              // lds_addr (bytes)
  g0[2] = (unsigned)(ga & 0xFFFFFFFFu);              // global_addr[31:0]
  g0[3] = (unsigned)((ga >> 32) & 0x1FFFFFFu)        // global_addr[56:32]
        | (2u << 30);                                // type=2 ("image")
  i32x8 g1;
  g1[0] = (1 << 16)                                  // data_size=2B
        | (1 << 20)                                  // pad_enable
        | (3 << 22)                                  // pad_interval: 16 DWORDs
        | (3 << 25);                                 // pad_amount: 4 DWORDs
  g1[1] = (int)((tdim0 & 0xFFFFu) << 16);            // tensor_dim0[15:0]
  g1[2] = (int)(((tdim0 >> 16) & 0xFFFFu) | ((tdim1 & 0xFFFFu) << 16));
  g1[3] = (int)(((tdim1 >> 16) & 0xFFFFu) | ((unsigned)BT_K << 16));  // tile_dim0=32
  g1[4] = BT_ROWS;                                   // tile_dim1=64, tile_dim2=0
  g1[5] = (int)src_stride;                           // tensor_dim0_stride[31:0]
  g1[6] = 0;
  g1[7] = 0;
  i32x4 z4 = {0, 0, 0, 0};                           // groups 2/3 unused (2D)
  i32x8 z8 = {0, 0, 0, 0, 0, 0, 0, 0};               // extra descriptor words
  __builtin_amdgcn_tensor_load_to_lds(g0, g1, z4, z4, z8, 0);
}

// Row-wise reductions within 16-lane halves (f32 C/D layout).
static __device__ inline float half_max(float v) {
  v = fmaxf(v, __shfl_xor(v, 1, 32));
  v = fmaxf(v, __shfl_xor(v, 2, 32));
  v = fmaxf(v, __shfl_xor(v, 4, 32));
  v = fmaxf(v, __shfl_xor(v, 8, 32));
  return v;
}
static __device__ inline float half_sum(float v) {
  v += __shfl_xor(v, 1, 32);
  v += __shfl_xor(v, 2, 32);
  v += __shfl_xor(v, 4, 32);
  v += __shfl_xor(v, 8, 32);
  return v;
}

// ---------------- Kernel 0: fp32 -> bf16 bulk convert ----------------
__global__ __launch_bounds__(256) void cvt_bf16_kernel(
    const float* __restrict__ src, __bf16* __restrict__ dst, int count8) {
  const int idx    = blockIdx.x * blockDim.x + threadIdx.x;
  const int stride = gridDim.x * blockDim.x;
  for (int i = idx; i < count8; i += stride) {
    v8f f = *(const v8f*)(src + (size_t)i * 8);
    v8bf b;
#pragma unroll
    for (int j = 0; j < 8; ++j) b[j] = (__bf16)f[j];
    *(v8bf*)(dst + (size_t)i * 8) = b;
  }
}

// ---------------- TDM-staged GEMM body (shared by kernels 1 and 3) ------
// Block: 8 waves, 128 rows x 64 cols.  All waves share the B strip, which is
// ping-pong DMA'd into LDS by wave 0 while everyone computes.
#define GEMM_PROLOGUE_AND_LOOP(Asrc, Bsrc, NSRC)                               \
  __shared__ __bf16 btile[2][BT_ELEMS];                                        \
  const int lane = threadIdx.x & 31;                                           \
  const int warp = threadIdx.x >> 5;                                           \
  const int row0 = (blockIdx.x & 31) * 128 + warp * 16;                        \
  const int col0 = (blockIdx.x >> 5) * 64;                                     \
  v8f c0 = {}, c1 = {}, c2 = {}, c3 = {};                                      \
  if (warp == 0) {                                                             \
    tdm_load_tile(Bsrc + (size_t)col0 * E_,                                    \
                  (unsigned)(uintptr_t)&btile[0][0], E_, NSRC, E_);            \
    __builtin_amdgcn_s_wait_tensorcnt(0);                                      \
  }                                                                            \
  __syncthreads();                                                             \
  v16bf a = load_a_bf16(Asrc, E_, row0, 0, lane);                              \
  int cur = 0;                                                                 \
  for (int k = 0; k < E_; k += 32) {                                           \
    if (warp == 0 && k + 32 < E_)                                              \
      tdm_load_tile(Bsrc + (size_t)col0 * E_ + k + 32,                         \
                    (unsigned)(uintptr_t)&btile[cur ^ 1][0], E_, NSRC, E_);    \
    v16bf an = a;                                                              \
    if (k + 32 < E_) an = load_a_bf16(Asrc, E_, row0, k + 32, lane);           \
    const __bf16* lb = &btile[cur][0];                                         \
    c0 = wmma_bf16(a, lds_bt(lb, 0,  lane), c0);                               \
    c1 = wmma_bf16(a, lds_bt(lb, 16, lane), c1);                               \
    c2 = wmma_bf16(a, lds_bt(lb, 32, lane), c2);                               \
    c3 = wmma_bf16(a, lds_bt(lb, 48, lane), c3);                               \
    a = an;                                                                    \
    if (warp == 0) __builtin_amdgcn_s_wait_tensorcnt(0);                       \
    __syncthreads();                                                           \
    cur ^= 1;                                                                  \
  }

// ---------------- Kernel 1: QKV projection ----------------
// C[4096,3072] = Xb @ Wqkvb^T ; scatter into Q[bh,n,d], K[bh,n,d], Vt[bh,d,n].
__global__ __launch_bounds__(256) void qkv_gemm_kernel(
    const __bf16* __restrict__ Xb, const __bf16* __restrict__ Wqb,
    __bf16* __restrict__ Qb, __bf16* __restrict__ Kb, __bf16* __restrict__ Vt) {
  GEMM_PROLOGUE_AND_LOOP(Xb, Wqb, 3 * E_)

  const int n = lane & 15, g = lane >> 4;
  v8f cs[4] = {c0, c1, c2, c3};
#pragma unroll
  for (int t = 0; t < 4; ++t) {
    const int j    = col0 + t * 16 + n;   // feature in [0,3072)
    const int part = j >> 10;             // 0=Q 1=K 2=V
    const int f    = j & 1023;
    const int h    = f >> 6;
    const int d    = f & 63;
#pragma unroll
    for (int r = 0; r < 8; ++r) {
      const int m  = row0 + r + g * 8;    // token in [0,4096)
      const int bh = (m >> 10) * H_ + h;
      const int nn = m & 1023;
      const __bf16 v = (__bf16)cs[t][r];
      if (part == 0)      Qb[((size_t)bh * N_ + nn) * D_ + d] = v;
      else if (part == 1) Kb[((size_t)bh * N_ + nn) * D_ + d] = v;
      else                Vt[((size_t)bh * D_ + d) * N_ + nn] = v;
    }
  }
}

// ---------------- Kernel 2: flash attention ----------------
__global__ __launch_bounds__(256) void attn_flash_kernel(
    const __bf16* __restrict__ Qb, const __bf16* __restrict__ Kb,
    const __bf16* __restrict__ Vt, const unsigned char* __restrict__ pad_mask,
    __bf16* __restrict__ Ob) {
  __shared__ __bf16 plds[8][16][48];
  const int lane = threadIdx.x & 31;
  const int warp = threadIdx.x >> 5;
  const int wid  = blockIdx.x * 8 + warp;       // 4096 waves
  const int bh   = wid >> 6;
  const int q0   = (wid & 63) * 16;
  const int n = lane & 15, g = lane >> 4;

  const __bf16* Qbh = Qb + (size_t)bh * N_ * D_;
  const __bf16* Kbh = Kb + (size_t)bh * N_ * D_;
  const __bf16* Vbh = Vt + (size_t)bh * D_ * N_;

  const v16bf qa0 = load_a_bf16(Qbh, D_, q0, 0, lane);
  const v16bf qa1 = load_a_bf16(Qbh, D_, q0, 32, lane);

  v8f acc0 = {}, acc1 = {}, acc2 = {}, acc3 = {};
  float mrow[8], lrow[8];
#pragma unroll
  for (int r = 0; r < 8; ++r) { mrow[r] = -3.0e38f; lrow[r] = 0.0f; }

  const int kend = q0 + 16;                     // causal horizon
  for (int k0 = 0; k0 < kend; k0 += 32) {
    // K fragments first (needed by S), V fragments stay in flight to the PV.
    const v16bf kb00 = load_bt_bf16(Kbh, D_, k0,      0,  lane);
    const v16bf kb01 = load_bt_bf16(Kbh, D_, k0,      32, lane);
    const v16bf kb10 = load_bt_bf16(Kbh, D_, k0 + 16, 0,  lane);
    const v16bf kb11 = load_bt_bf16(Kbh, D_, k0 + 16, 32, lane);
    const v16bf vb0  = load_bt_bf16(Vbh, N_, 0,  k0, lane);
    const v16bf vb1  = load_bt_bf16(Vbh, N_, 16, k0, lane);
    const v16bf vb2  = load_bt_bf16(Vbh, N_, 32, k0, lane);
    const v16bf vb3  = load_bt_bf16(Vbh, N_, 48, k0, lane);

    v8f s0 = {}, s1 = {};
    s0 = wmma_bf16(qa0, kb00, s0);
    s0 = wmma_bf16(qa1, kb01, s0);
    s1 = wmma_bf16(qa0, kb10, s1);
    s1 = wmma_bf16(qa1, kb11, s1);

    const int kc0 = k0 + n;
    const int kc1 = k0 + 16 + n;
    const int i0  = (kc0 == 0) ? 0 : (kc0 - 1);
    const bool ok0 = (kc0 == 0) | (pad_mask[(size_t)bh * (N_ - 1) + i0] != 0);
    const bool ok1 = (pad_mask[(size_t)bh * (N_ - 1) + (kc1 - 1)] != 0);

#pragma unroll
    for (int r = 0; r < 8; ++r) {
      const int row = q0 + r + g * 8;
      float v0 = s0[r] * 0.25f;                 // scale = H^-0.5 (faithful)
      float v1 = s1[r] * 0.25f;
      v0 = ((kc0 <= row) & ok0) ? v0 : -3.0e38f;
      v1 = ((kc1 <= row) & ok1) ? v1 : -3.0e38f;
      const float tmax  = half_max(fmaxf(v0, v1));
      const float mnew  = fmaxf(mrow[r], tmax);
      const float alpha = __expf(mrow[r] - mnew);
      const float p0    = __expf(v0 - mnew);
      const float p1    = __expf(v1 - mnew);
      lrow[r] = lrow[r] * alpha + half_sum(p0 + p1);
      mrow[r] = mnew;
      acc0[r] *= alpha; acc1[r] *= alpha; acc2[r] *= alpha; acc3[r] *= alpha;
      plds[warp][r + g * 8][n]      = (__bf16)p0;
      plds[warp][r + g * 8][16 + n] = (__bf16)p1;
    }

    asm volatile("s_wait_dscnt 0" ::: "memory");
    const __bf16* pb = &plds[warp][n][0];
    v8bf lo = *(const v8bf*)(pb + g * 16);
    v8bf hi = *(const v8bf*)(pb + g * 16 + 8);
    v16bf pa;
#pragma unroll
    for (int i = 0; i < 8; ++i) { pa[i] = lo[i]; pa[i + 8] = hi[i]; }
    asm volatile("" ::: "memory");

    acc0 = wmma_bf16(pa, vb0, acc0);
    acc1 = wmma_bf16(pa, vb1, acc1);
    acc2 = wmma_bf16(pa, vb2, acc2);
    acc3 = wmma_bf16(pa, vb3, acc3);
  }

#pragma unroll
  for (int r = 0; r < 8; ++r) {
    const float inv = 1.0f / lrow[r];
    __bf16* op = Ob + ((size_t)bh * N_ + q0 + r + g * 8) * D_ + n;
    op[0]  = (__bf16)(acc0[r] * inv);
    op[16] = (__bf16)(acc1[r] * inv);
    op[32] = (__bf16)(acc2[r] * inv);
    op[48] = (__bf16)(acc3[r] * inv);
  }
}

// ---------------- Kernel 3: output projection ----------------
__global__ __launch_bounds__(256) void proj_gemm_kernel(
    const __bf16* __restrict__ Ab, const __bf16* __restrict__ Wpb,
    float* __restrict__ Out) {
  GEMM_PROLOGUE_AND_LOOP(Ab, Wpb, E_)

  const int n = lane & 15, g = lane >> 4;
  v8f cs[4] = {c0, c1, c2, c3};
#pragma unroll
  for (int t = 0; t < 4; ++t)
#pragma unroll
    for (int r = 0; r < 8; ++r)
      Out[(size_t)(row0 + r + g * 8) * E_ + col0 + t * 16 + n] = cs[t][r];
}

// ---------------- host launcher ----------------
extern "C" void kernel_launch(void* const* d_in, const int* in_sizes, int n_in,
                              void* d_out, int out_size, void* d_ws, size_t ws_size,
                              hipStream_t stream) {
  (void)in_sizes; (void)n_in; (void)out_size; (void)ws_size;
  const float* x           = (const float*)d_in[0];
  /* d_in[1] = causal mask: reproduced analytically, not read */
  const unsigned char* pad = (const unsigned char*)d_in[2];
  const float* w_qkv       = (const float*)d_in[3];
  const float* w_proj      = (const float*)d_in[4];
  float* out               = (float*)d_out;

  const size_t nelem = (size_t)BH_ * N_ * D_;   // 4M elements
  __bf16* Xb  = (__bf16*)d_ws;                  // 8 MB
  __bf16* Wqb = Xb  + (size_t)B_ * N_ * E_;     // 6 MB
  __bf16* Wpb = Wqb + (size_t)3 * E_ * E_;      // 2 MB
  __bf16* Qb  = Wpb + (size_t)E_ * E_;          // 8 MB
  __bf16* Kb  = Qb  + nelem;                    // 8 MB
  __bf16* Vt  = Kb  + nelem;                    // 8 MB (V stored [bh, d, n])
  __bf16* Ob  = Vt  + nelem;                    // 8 MB  -> 48 MB total ws

  cvt_bf16_kernel<<<2048, 256, 0, stream>>>(x,      Xb,  (int)((size_t)B_ * N_ * E_ / 8));
  cvt_bf16_kernel<<<1536, 256, 0, stream>>>(w_qkv,  Wqb, (int)((size_t)3 * E_ * E_ / 8));
  cvt_bf16_kernel<<<512,  256, 0, stream>>>(w_proj, Wpb, (int)((size_t)E_ * E_ / 8));
  qkv_gemm_kernel<<<1536, 256, 0, stream>>>(Xb, Wqb, Qb, Kb, Vt);     // 32x48 blocks
  attn_flash_kernel<<<512, 256, 0, stream>>>(Qb, Kb, Vt, pad, Ob);
  proj_gemm_kernel<<<512, 256, 0, stream>>>(Ob, Wpb, out);            // 32x16 blocks
}
